// DecoderLSTM_42846593745222
// MI455X (gfx1250) — matrix operand order repaired and checked by reading, not compiled, not run on hardware
//
#include <hip/hip_runtime.h>
#include <hip/hip_bf16.h>

// ---------------------------------------------------------------------------
// DecoderLSTM on MI455X (gfx1250, wave32, WMMA, TDM, 320KB LDS)
//
// 515 GFLOP total; weights = 15.7MB bf16 (L2-resident). Latency-bound:
// 256 sequential steps x 3 dependent GEMM phases. Persistent kernel
// (64 blocks x 256 thr), device-scope grid barrier between phases.
// Loop-invariant per-block weight slabs (160KB + 64KB) are TDM-staged into
// LDS once, frag-major; per-step A/B fragments are double-buffered one kt
// ahead (global + DS) so v_wmma issue is not gated on load latency.
// ---------------------------------------------------------------------------

#define UNITS 1024
#define TOKEN 512
#define TSEQ  256
#define BATCH 128
#define NBLK  64

#define LDSA_ELEMS (32 * 5 * 32 * 16)   // [kt=32][j=5][lane=32][e=16] = 81920 bf16 (160KB)
#define LDSC_ELEMS (16 * 4 * 32 * 16)   // [kt=16][g=4][lane=32][e=16] = 32768 bf16 (64KB)
#define LDS_BYTES  ((LDSA_ELEMS + LDSC_ELEMS) * 2)   // 224KB of the 320KB WGP LDS

typedef __attribute__((ext_vector_type(16))) __bf16 v16bf;
typedef __attribute__((ext_vector_type(8)))  __bf16 v8bf;
typedef __attribute__((ext_vector_type(8)))  float  v8f;
typedef __attribute__((ext_vector_type(4)))  unsigned v4u;
typedef __attribute__((ext_vector_type(8)))  int      v8i;
typedef __attribute__((ext_vector_type(4)))  int      v4i;

__device__ __forceinline__ v8f wmma_bf16(v16bf a, v16bf b, v8f c) {
  return __builtin_amdgcn_wmma_f32_16x16x32_bf16(
      false, a, false, b, (short)0, c, false, false);
}

// A-fragment (16x32 MxK) from row-major bf16, ld elements per row.
// Lanes 0-15: row M=lane, K {0..15}; lanes 16-31: row M=lane-16,
// K {16..31}. Two contiguous 16B loads per lane.
__device__ __forceinline__ v16bf load_afrag(const __bf16* base, int ld,
                                            int m0, int k0, int lane) {
  const int half = lane >> 4;
  const int r    = lane & 15;
  const __bf16* p = base + (long)(m0 + r) * ld + k0 + half * 16;
  v8bf lo = *(const v8bf*)(p);
  v8bf hi = *(const v8bf*)(p + 8);
  v16bf out;
#pragma unroll
  for (int i = 0; i < 8; ++i) { out[i] = lo[i]; out[i + 8] = hi[i]; }
  return out;
}

// B-fragment from frag-major LDS slab: fragment f occupies 32 lanes x 16 bf16
// contiguous; each lane reads its own 32B chunk (2x ds_load_b128).
__device__ __forceinline__ v16bf lds_bfrag(const __bf16* lds, int f, int lane) {
  return *(const v16bf*)(lds + ((long)f * 32 + lane) * 16);
}

// B-fragment from column-major global weights [N][Kdim] (phase B only).
__device__ __forceinline__ v16bf load_bfrag(const __bf16* Wt, int Kdim,
                                            int n0, int k0, int lane) {
  const int half = lane >> 4;
  const int c    = lane & 15;
  const __bf16* p = Wt + (long)(n0 + c) * Kdim + k0 + half * 16;
  return *(const v16bf*)p;
}

__device__ __forceinline__ float sigmf(float x) {
  return 1.0f / (1.0f + __expf(-x));
}

// 1-D TDM copy global->LDS: nbytes (multiple of 8, tile_dim0 < 64K elems) as a
// 1-row tile of 8-byte elements. D# layout per CDNA5 ISA ch.8 (group0/group1).
__device__ __forceinline__ void tdm_load_1d(unsigned lds_addr, const void* gptr,
                                            unsigned nbytes) {
  unsigned long long ga = (unsigned long long)gptr;
  unsigned n8 = nbytes >> 3;
  v4u g0 = { 1u,                                   // count=1, user desc, no gather
             lds_addr,                             // lds_addr[31:0]
             (unsigned)(ga & 0xffffffffu),         // global_addr[31:0]
             (unsigned)((ga >> 32) & 0x01ffffffu)  // global_addr[56:32]
                 | (2u << 30) };                   // type=2 ("image")
  v8i g1 = { (int)(3u << 16),        // workgroup_mask=0, data_size=3 (8B)
             (int)(n8 << 16),        // tensor_dim0[15:0]
             (int)(n8 >> 16),        // tensor_dim0[31:16]; tensor_dim1=0
             (int)(n8 << 16),        // tile_dim0 = n8
             0,                      // tile_dim1=0 (unused), tile_dim2=0
             (int)n8,                // tensor_dim0_stride lo
             0, 0 };
  v4i z4 = { 0, 0, 0, 0 };
#if defined(__clang_major__) && (__clang_major__ >= 23)
  v8i z8 = { 0, 0, 0, 0, 0, 0, 0, 0 };
  __builtin_amdgcn_tensor_load_to_lds(g0, g1, z4, z4, z8, 0);
#else
  __builtin_amdgcn_tensor_load_to_lds(g0, g1, z4, z4, 0);
#endif
}

// Device-wide sense-reversal barrier on two L2-resident counters.
__device__ __forceinline__ void grid_barrier(unsigned* bar) {
  __builtin_amdgcn_fence(__ATOMIC_RELEASE, "agent");
  __syncthreads();
  if (threadIdx.x == 0) {
    unsigned gen  = __hip_atomic_load(&bar[1], __ATOMIC_RELAXED, __HIP_MEMORY_SCOPE_AGENT);
    unsigned prev = __hip_atomic_fetch_add(&bar[0], 1u, __ATOMIC_RELAXED, __HIP_MEMORY_SCOPE_AGENT);
    if (prev == NBLK - 1) {
      __hip_atomic_store(&bar[0], 0u, __ATOMIC_RELAXED, __HIP_MEMORY_SCOPE_AGENT);
      __hip_atomic_fetch_add(&bar[1], 1u, __ATOMIC_RELAXED, __HIP_MEMORY_SCOPE_AGENT);
    } else {
      while (__hip_atomic_load(&bar[1], __ATOMIC_RELAXED, __HIP_MEMORY_SCOPE_AGENT) == gen)
        __builtin_amdgcn_s_sleep(2);
    }
  }
  __syncthreads();
  __builtin_amdgcn_fence(__ATOMIC_ACQUIRE, "agent");
}

// ---------------------------------------------------------------------------
// Prep: zero barrier; pack weights bf16 in per-block TDM/LDS frag-major order;
// WtB column-major; h0 = bf16(s_0). Gate order everywhere: [i, f, o, g].
// ---------------------------------------------------------------------------
__global__ void prep_kernel(const float* __restrict__ Wi, const float* __restrict__ Wf,
                            const float* __restrict__ Wo, const float* __restrict__ Wg,
                            const float* __restrict__ Wd1, const float* __restrict__ Wd2,
                            const float* __restrict__ s0,
                            unsigned short* WtAp_u, unsigned short* WtB_u,
                            unsigned short* WtCp_u, unsigned short* hbuf_u,
                            unsigned* bar) {
  __bf16* WtAp = (__bf16*)WtAp_u;
  __bf16* WtB  = (__bf16*)WtB_u;
  __bf16* WtCp = (__bf16*)WtCp_u;
  __bf16* hbuf = (__bf16*)hbuf_u;

  const long idx    = (long)blockIdx.x * blockDim.x + threadIdx.x;
  const long stride = (long)gridDim.x * blockDim.x;
  if (idx < 2) bar[idx] = 0u;

  const long nA = (long)NBLK * LDSA_ELEMS;
  const long nB = 512L * 1024;
  const long nC = (long)NBLK * LDSC_ELEMS;
  const long nH = 128L * 1024;

  for (long i = idx; i < nA; i += stride) {
    long nb = i / LDSA_ELEMS, r = i - nb * LDSA_ELEMS;
    long kt = r / 2560, r2 = r - kt * 2560;
    long j = r2 >> 9, r3 = r2 & 511;
    long lane = r3 >> 4, e = r3 & 15;
    long col = (nb * 5 + j) * 16 + (lane & 15);
    long k   = kt * 32 + (lane >> 4) * 16 + e;
    float v;
    if (col < 1024) {
      v = Wd1[k * 1024 + col];
    } else {
      long g = (col - 1024) >> 10, c = (col - 1024) & 1023;
      const float* W = (g == 0) ? Wi : (g == 1) ? Wf : (g == 2) ? Wo : Wg;
      v = W[(512 + k) * 1024 + c];    // h-rows of the concat weight
    }
    WtAp[i] = (__bf16)v;
  }
  for (long i = idx; i < nB; i += stride) {
    long n = i >> 10, k = i & 1023;
    WtB[n * 1024 + k] = (__bf16)Wd2[k * 512 + n];
  }
  for (long i = idx; i < nC; i += stride) {
    long nb = i >> 15, r = i & 32767;
    long kt = r >> 11, r2 = r & 2047;
    long g = r2 >> 9, r3 = r2 & 511;
    long lane = r3 >> 4, e = r3 & 15;
    long c = nb * 16 + (lane & 15);
    long k = kt * 32 + (lane >> 4) * 16 + e;
    const float* W = (g == 0) ? Wi : (g == 1) ? Wf : (g == 2) ? Wo : Wg;
    WtCp[i] = (__bf16)W[k * 1024 + c];  // y-rows of the concat weight
  }
  for (long i = idx; i < nH; i += stride) hbuf[i] = (__bf16)s0[i];
}

// ---------------------------------------------------------------------------
// Persistent main kernel.
// ---------------------------------------------------------------------------
__global__ __launch_bounds__(256) void lstm_main(
    const unsigned short* __restrict__ WtAp_u, const unsigned short* __restrict__ WtB_u,
    const unsigned short* __restrict__ WtCp_u,
    unsigned short* hbuf_u, unsigned short* y1buf_u, unsigned short* ybuf_u,
    float* __restrict__ gpart,
    const float* __restrict__ c0,
    const float* __restrict__ bi, const float* __restrict__ bfv,
    const float* __restrict__ bo, const float* __restrict__ bg,
    const float* __restrict__ bd1, const float* __restrict__ bd2,
    float* __restrict__ out, unsigned* bar) {
  extern __shared__ __align__(16) char smem[];
  __bf16* ldsA = (__bf16*)smem;                       // 160KB
  __bf16* ldsC = (__bf16*)(smem + LDSA_ELEMS * 2);    // 64KB

  const __bf16* WtAp = (const __bf16*)WtAp_u;
  const __bf16* WtB  = (const __bf16*)WtB_u;
  const __bf16* WtCp = (const __bf16*)WtCp_u;
  __bf16* hbuf  = (__bf16*)hbuf_u;
  __bf16* y1buf = (__bf16*)y1buf_u;
  __bf16* ybuf  = (__bf16*)ybuf_u;

  const int tid  = threadIdx.x;
  const int lane = tid & 31;
  const int wave = tid >> 5;           // 8 waves; wave == M-tile in phases A/C
  const int blk  = blockIdx.x;
  const int gw   = blk * 8 + wave;
  const int half = lane >> 4;
  const int cL   = lane & 15;
  const int m0   = wave * 16;          // this wave's 16 batch rows

  // ---- One-time TDM stage of this block's loop-invariant weight slabs ----
  if (wave == 0) {
    tdm_load_1d((unsigned)(unsigned long long)(void*)ldsA,
                WtAp + (long)blk * LDSA_ELEMS, LDSA_ELEMS * 2);
    tdm_load_1d((unsigned)(unsigned long long)(void*)ldsC,
                WtCp + (long)blk * LDSC_ELEMS, LDSC_ELEMS * 2);
    __builtin_amdgcn_s_wait_tensorcnt(0);
  }
  __syncthreads();

  for (int t = 0; t < TSEQ; ++t) {
    // ---- Phase A: y1 = tanh(h*Wd1+bd1) and gate h-partials ----------------
    // [128 x 1024] x [1024 x 5120], A from global (pipelined), B from LDS
    // (double-buffered one kt ahead so s_wait_dscnt never blocks wmma issue).
    {
      v8f acc[5] = {};
      v16bf a = load_afrag(hbuf, UNITS, m0, 0, lane);
      v16bf bcur[5];
#pragma unroll
      for (int j = 0; j < 5; ++j) bcur[j] = lds_bfrag(ldsA, j, lane);
      for (int kt = 0; kt < 32; ++kt) {
        v16bf an = a;
        v16bf bnxt[5];
#pragma unroll
        for (int j = 0; j < 5; ++j) bnxt[j] = bcur[j];
        if (kt + 1 < 32) {
          an = load_afrag(hbuf, UNITS, m0, (kt + 1) * 32, lane);
#pragma unroll
          for (int j = 0; j < 5; ++j) bnxt[j] = lds_bfrag(ldsA, (kt + 1) * 5 + j, lane);
        }
#pragma unroll
        for (int j = 0; j < 5; ++j) acc[j] = wmma_bf16(a, bcur[j], acc[j]);
        a = an;
#pragma unroll
        for (int j = 0; j < 5; ++j) bcur[j] = bnxt[j];
      }
#pragma unroll
      for (int j = 0; j < 5; ++j) {
        const int n0  = (blk * 5 + j) * 16;
        const int col = n0 + cL;
        if (n0 < 1024) {
          const float bias = bd1[col];
#pragma unroll
          for (int r = 0; r < 8; ++r) {
            const int row = m0 + r + half * 8;
            y1buf[(long)row * UNITS + col] = (__bf16)tanhf(acc[j][r] + bias);
          }
        } else {
          const int gcol = col - 1024;   // [i|f|o|g] * 1024
#pragma unroll
          for (int r = 0; r < 8; ++r) {
            const int row = m0 + r + half * 8;
            gpart[(long)row * 4096 + gcol] = acc[j][r];
          }
        }
      }
    }
    grid_barrier(bar);

    // ---- Phase B: y = y1*Wd2 + bd2 -> d_out[b,t,:] and bf16 ybuf ----------
    if (gw < 256) {
      const int mt = gw >> 5, nt = gw & 31;
      const int mb0 = mt * 16, n0 = nt * 16;
      v8f acc = {};
      v16bf a = load_afrag(y1buf, UNITS, mb0, 0, lane);
      v16bf b = load_bfrag(WtB, 1024, n0, 0, lane);
      for (int kt = 0; kt < 32; ++kt) {
        v16bf an = a, bn = b;
        if (kt + 1 < 32) {
          an = load_afrag(y1buf, UNITS, mb0, (kt + 1) * 32, lane);
          bn = load_bfrag(WtB, 1024, n0, (kt + 1) * 32, lane);
        }
        acc = wmma_bf16(a, b, acc);
        a = an; b = bn;
      }
      const int col    = n0 + cL;
      const float bias = bd2[col];
#pragma unroll
      for (int r = 0; r < 8; ++r) {
        const int row = mb0 + r + half * 8;
        const float v = acc[r] + bias;
        out[(long)row * (TSEQ * TOKEN) + (long)t * TOKEN + col] = v;  // [B,T,TOKEN]
        ybuf[(long)row * TOKEN + col] = (__bf16)v;
      }
    }
    grid_barrier(bar);

    // ---- Phase C: gates = y*W_y + gpart + b ; h = o*tanh(f*c0+i*g) --------
    {
      const int n0 = blk * 16;            // nc = blk
      v8f acc[4] = {};
      v16bf a = load_afrag(ybuf, TOKEN, m0, 0, lane);
      v16bf bcur[4];
#pragma unroll
      for (int g = 0; g < 4; ++g) bcur[g] = lds_bfrag(ldsC, g, lane);
      for (int kt = 0; kt < 16; ++kt) {   // K = TOKEN = 512
        v16bf an = a;
        v16bf bnxt[4];
#pragma unroll
        for (int g = 0; g < 4; ++g) bnxt[g] = bcur[g];
        if (kt + 1 < 16) {
          an = load_afrag(ybuf, TOKEN, m0, (kt + 1) * 32, lane);
#pragma unroll
          for (int g = 0; g < 4; ++g) bnxt[g] = lds_bfrag(ldsC, (kt + 1) * 4 + g, lane);
        }
#pragma unroll
        for (int g = 0; g < 4; ++g) acc[g] = wmma_bf16(a, bcur[g], acc[g]);
        a = an;
#pragma unroll
        for (int g = 0; g < 4; ++g) bcur[g] = bnxt[g];
      }
      const int col = n0 + cL;
      const float Bi = bi[col], Bf = bfv[col], Bo = bo[col], Bg = bg[col];
#pragma unroll
      for (int r = 0; r < 8; ++r) {
        const int row = m0 + r + half * 8;
        const float* gp = gpart + (long)row * 4096;
        const float gi = sigmf(acc[0][r] + gp[col]        + Bi);
        const float gf = sigmf(acc[1][r] + gp[1024 + col] + Bf);
        const float go = sigmf(acc[2][r] + gp[2048 + col] + Bo);
        const float gg = tanhf(acc[3][r] + gp[3072 + col] + Bg);
        const float cn = gf * c0[(long)row * UNITS + col] + gi * gg;  // c_0 frozen
        hbuf[(long)row * UNITS + col] = (__bf16)(go * tanhf(cn));
      }
    }
    grid_barrier(bar);   // h visible before next step's phase A
  }
}

// ---------------------------------------------------------------------------
extern "C" void kernel_launch(void* const* d_in, const int* in_sizes, int n_in,
                              void* d_out, int out_size, void* d_ws, size_t ws_size,
                              hipStream_t stream) {
  const float* s0  = (const float*)d_in[0];
  const float* c0  = (const float*)d_in[1];
  const float* Wi  = (const float*)d_in[2];
  const float* Wf  = (const float*)d_in[3];
  const float* Wg  = (const float*)d_in[4];
  const float* Wo  = (const float*)d_in[5];
  const float* bi  = (const float*)d_in[6];
  const float* bf  = (const float*)d_in[7];
  const float* bg  = (const float*)d_in[8];
  const float* bo  = (const float*)d_in[9];
  const float* Wd1 = (const float*)d_in[10];
  const float* bd1 = (const float*)d_in[11];
  const float* Wd2 = (const float*)d_in[12];
  const float* bd2 = (const float*)d_in[13];
  float* out = (float*)d_out;

  // Workspace layout (bytes), 256B-aligned segments. Total ~18.5 MB.
  char* ws = (char*)d_ws;
  size_t off = 0;
  unsigned* bar = (unsigned*)(ws + off);              off += 256;
  unsigned short* WtAp = (unsigned short*)(ws + off); off += (size_t)NBLK * LDSA_ELEMS * 2;
  unsigned short* WtB  = (unsigned short*)(ws + off); off += (size_t)512 * 1024 * 2;
  unsigned short* WtCp = (unsigned short*)(ws + off); off += (size_t)NBLK * LDSC_ELEMS * 2;
  unsigned short* hbuf = (unsigned short*)(ws + off); off += (size_t)128 * 1024 * 2;
  unsigned short* y1b  = (unsigned short*)(ws + off); off += (size_t)128 * 1024 * 2;
  unsigned short* yb   = (unsigned short*)(ws + off); off += (size_t)128 * 512 * 2;
  float* gpart = (float*)(ws + off);                  off += (size_t)128 * 4096 * 4;

  prep_kernel<<<2048, 256, 0, stream>>>(Wi, Wf, Wo, Wg, Wd1, Wd2, s0,
                                        WtAp, WtB, WtCp, hbuf, bar);

  lstm_main<<<NBLK, 256, LDS_BYTES, stream>>>(WtAp, WtB, WtCp, hbuf, y1b, yb,
                                              gpart, c0, bi, bf, bo, bg,
                                              bd1, bd2, out, bar);
}
